// SelfAttentionUnit_5317169512778
// MI455X (gfx1250) — matrix operand order repaired and checked by reading, not compiled, not run on hardware
//
#include <hip/hip_runtime.h>
#include <hip/hip_bf16.h>

#define B_   8
#define CIN  67
#define C2   128
#define C_   64
#define N_   16384
#define KP   96          // CIN zero-padded to multiple of 32
#define EPS_ 1e-5f

typedef __bf16 bf16_t;
typedef __attribute__((ext_vector_type(16))) __bf16 v16bf;
typedef __attribute__((ext_vector_type(8)))  float  v8f;

__device__ __forceinline__ v8f wmma_bf16(v16bf a, v16bf b, v8f c) {
  // D = A(16x32 bf16) * B(32x16 bf16) + C(16x16 f32)
  return __builtin_amdgcn_wmma_f32_16x16x32_bf16(false, a, false, b, (short)0, c, false, false);
}

// 32-bit LDS byte offset of a generic pointer that points into LDS.
__device__ __forceinline__ unsigned lds_offset(const void* p) {
  return (unsigned)(size_t)(__attribute__((address_space(3))) const void*)p;
}

// Async copy of 16 bytes per lane: global -> LDS, tracked by ASYNCcnt.
__device__ __forceinline__ void async_copy_b128(unsigned lds_off, const void* gaddr) {
  asm volatile("global_load_async_to_lds_b128 %0, %1, off"
               :: "v"(lds_off), "v"((unsigned long long)(size_t)gaddr)
               : "memory");
}
__device__ __forceinline__ void async_wait0() {
  asm volatile("s_wait_asynccnt 0x0" ::: "memory");
}

// ---- Fragment loaders (wave32; A and B both use contiguous k = half*16 + 0..15) ----

// A fragment: element (m, k), bf16 source, k contiguous, row stride ld (elems).
// lane: m = lane&15 ; K = half*16 + 0..15 contiguous.
__device__ __forceinline__ v16bf load_a_bf16(const bf16_t* mat, int ld, int m, int kbase, int half) {
  const bf16_t* row = mat + (size_t)m * ld + kbase;
  union { v16bf v; uint4 q[2]; } u;
  u.q[0] = *(const uint4*)(row + half * 16);
  u.q[1] = *(const uint4*)(row + half * 16 + 8);
  return u.v;
}

// B fragment: element (k, n) stored at matT[n*ld + k] (bf16, k contiguous).
// lane: n = lane&15 ; K = half*16 + 0..15 contiguous.
__device__ __forceinline__ v16bf load_b_bf16(const bf16_t* matT, int ld, int n, int kbase, int half) {
  const bf16_t* p = matT + (size_t)n * ld + kbase + half * 16;
  union { v16bf v; uint4 q[2]; } u;
  u.q[0] = *(const uint4*)(p);
  u.q[1] = *(const uint4*)(p + 8);
  return u.v;
}

// A fragment from f32 row-major (row stride ld), guarded K < klim, cvt->bf16.
__device__ __forceinline__ v16bf load_a_f32(const float* mat, int ld, int m, int kbase, int klim, int half) {
  v16bf a;
  const float* row = mat + (size_t)m * ld;
#pragma unroll
  for (int i = 0; i < 8; ++i) {
    int k0 = half * 16 + 2 * i;
    int kk = kbase + k0;
    float f0 = (kk     < klim) ? row[kk]     : 0.0f;
    float f1 = (kk + 1 < klim) ? row[kk + 1] : 0.0f;
    a[2 * i]     = (bf16_t)f0;
    a[2 * i + 1] = (bf16_t)f1;
  }
  return a;
}

// B fragment from f32, element (k, n) at matT[n*ld + k], k contiguous.
__device__ __forceinline__ v16bf load_b_f32_contig(const float* matT, int ld, int n, int kbase, int half) {
  const float* p = matT + (size_t)n * ld + kbase + half * 16;
  v16bf b;
#pragma unroll
  for (int j = 0; j < 16; ++j) b[j] = (bf16_t)p[j];
  return b;
}

// ---- Stage P: pad+convert weights to bf16 [C2][KP]; fuse BN into scale/shift ----
// sc/sh layout: [q: 0..127][k: 128..255][v: 256..383][f: 384..447]
__global__ void prep_kernel(const float* __restrict__ Wq, const float* __restrict__ Wk,
                            const float* __restrict__ Wv,
                            const float* __restrict__ bnq, const float* __restrict__ bnk,
                            const float* __restrict__ bnv, const float* __restrict__ bnf,
                            bf16_t* __restrict__ Wqp, bf16_t* __restrict__ Wkp,
                            bf16_t* __restrict__ Wvp,
                            float* __restrict__ sc, float* __restrict__ sh) {
  const int WTOT = C2 * KP;
  int i = blockIdx.x * blockDim.x + threadIdx.x;
  if (i < 3 * WTOT) {
    int wsel = i / WTOT, r = i % WTOT;
    int m = r / KP, k = r % KP;
    const float* W = (wsel == 0) ? Wq : (wsel == 1) ? Wk : Wv;
    bf16_t* Wp     = (wsel == 0) ? Wqp : (wsel == 1) ? Wkp : Wvp;
    Wp[r] = (bf16_t)((k < CIN) ? W[m * CIN + k] : 0.0f);
  }
  int j = i - 3 * WTOT;
  if (j >= 0 && j < 3 * C2 + C_) {
    const float* bn; int ch, ld;
    if      (j <     C2) { bn = bnq; ch = j;          ld = C2; }
    else if (j < 2 * C2) { bn = bnk; ch = j - C2;     ld = C2; }
    else if (j < 3 * C2) { bn = bnv; ch = j - 2 * C2; ld = C2; }
    else                 { bn = bnf; ch = j - 3 * C2; ld = C_; }
    float g = bn[ch], be = bn[ld + ch], mn = bn[2 * ld + ch], vr = bn[3 * ld + ch];
    float s = g * rsqrtf(vr + EPS_);
    sc[j] = s;
    sh[j] = be - mn * s;
  }
}

// ---- Stage 0: zero the kv accumulation buffer (ws is poisoned) ----
__global__ void zero_kernel(float* __restrict__ p, int n) {
  int i = blockIdx.x * blockDim.x + threadIdx.x;
  if (i < n) p[i] = 0.0f;
}

// ---- Stage 1: q/k/v = ReLU(BN(W*x)); x tile staged in LDS, bf16 outputs ----
// grid (N/16, B), 256 thr = 8 waves; wave w owns output-channel tile c0=w*16.
__global__ __launch_bounds__(256) void qkv_kernel(
    const float* __restrict__ x,
    const bf16_t* __restrict__ Wqp, const bf16_t* __restrict__ Wkp, const bf16_t* __restrict__ Wvp,
    const float* __restrict__ sc, const float* __restrict__ sh,
    bf16_t* __restrict__ kbuf, bf16_t* __restrict__ vbuf, bf16_t* __restrict__ qT) {
  __shared__ uint4 lds4[16 * KP * 2 / 16];     // xT: 16 n-rows x KP k (bf16) = 3 KB
  bf16_t* xT = (bf16_t*)lds4;
  const int b  = blockIdx.y;
  const int n0 = blockIdx.x * 16;
  const int t  = threadIdx.x;

  // cooperative, transposing stage: xT[n][c] = x[b][c][n0+n] (guard c < CIN; cvt f32->bf16)
#pragma unroll
  for (int i = 0; i < 6; ++i) {
    int idx = t + i * 256;                     // 0..1535 over [c=96][n=16]
    int c = idx >> 4, j = idx & 15;
    float f = (c < CIN) ? x[((size_t)b * CIN + c) * N_ + n0 + j] : 0.0f;
    xT[j * KP + c] = (bf16_t)f;
  }
  __syncthreads();

  const int wave = t >> 5, lane = t & 31;
  const int half = lane >> 4, lm = lane & 15;
  const int c0 = wave * 16;

  v8f accq = {}, acck = {}, accv = {};
#pragma unroll
  for (int kb = 0; kb < KP; kb += 32) {
    v16bf bx = load_b_bf16(xT, KP, lm, kb, half);   // ds_load_b128 x2
    v16bf aq = load_a_bf16(Wqp, KP, c0 + lm, kb, half);
    v16bf ak = load_a_bf16(Wkp, KP, c0 + lm, kb, half);
    v16bf av = load_a_bf16(Wvp, KP, c0 + lm, kb, half);
    accq = wmma_bf16(aq, bx, accq);
    acck = wmma_bf16(ak, bx, acck);
    accv = wmma_bf16(av, bx, accv);
  }
  // D layout: n = n0+lm (lane col), channel = c0 + 8*half + i (VGPR row)
  const int n = n0 + lm;
  union { uint4 q; bf16_t h[8]; } qs;
#pragma unroll
  for (int i = 0; i < 8; ++i) {
    int ch = c0 + 8 * half + i;
    qs.h[i] = (bf16_t)fmaxf(fmaf(accq[i], sc[ch], sh[ch]), 0.0f);
    kbuf[((size_t)b * C2 + ch) * N_ + n] =
        (bf16_t)fmaxf(fmaf(acck[i], sc[C2 + ch], sh[C2 + ch]), 0.0f);
    vbuf[((size_t)b * C2 + ch) * N_ + n] =
        (bf16_t)fmaxf(fmaf(accv[i], sc[2 * C2 + ch], sh[2 * C2 + ch]), 0.0f);
  }
  // q stored transposed [B][N][C2]: 8 consecutive channels -> one b128 store
  *(uint4*)&qT[((size_t)b * N_ + n) * C2 + c0 + 8 * half] = qs.q;
}

// ---- Stage 2: kv[b] += k * v^T over a 256-wide N-chunk (split-K, f32 atomics) ----
// grid (64, B): v-chunk [128][256] async-staged once in LDS (64 KB), shared by 8 waves.
__global__ __launch_bounds__(256) void kv_kernel(
    const bf16_t* __restrict__ kbuf, const bf16_t* __restrict__ vbuf,
    float* __restrict__ kv) {
  __shared__ uint4 lds4[4096];                 // vt: [d=128][k=256] bf16 = 64 KB
  bf16_t* vt = (bf16_t*)lds4;
  const int b    = blockIdx.y;
  const int nbeg = blockIdx.x * 256;
  const int t    = threadIdx.x;
  const bf16_t* kb = kbuf + (size_t)b * C2 * N_;
  const bf16_t* vb = vbuf + (size_t)b * C2 * N_ + nbeg;

  // cooperative async stage: 4096 b128 chunks, 16 per thread (no VGPR round-trip)
  const unsigned lbase = lds_offset(lds4);
#pragma unroll
  for (int i = 0; i < 16; ++i) {
    int id = t + i * 256;                      // id = d*32 + c8
    int d = id >> 5, c8 = id & 31;
    async_copy_b128(lbase + (unsigned)id * 16, vb + (size_t)d * N_ + c8 * 8);
  }
  async_wait0();                               // ASYNCcnt == 0 for this wave
  __syncthreads();                             // all waves' chunks visible

  const int wave = t >> 5, lane = t & 31;
  const int half = lane >> 4, lm = lane & 15;
  const int c0 = wave * 16;

  v8f acc[8];
#pragma unroll
  for (int d = 0; d < 8; ++d) acc[d] = (v8f){};

#pragma unroll 2
  for (int nk = 0; nk < 256; nk += 32) {
    v16bf af = load_a_bf16(kb, N_, c0 + lm, nbeg + nk, half);  // A = k (m=c, K=n)
#pragma unroll
    for (int dt = 0; dt < 8; ++dt) {
      v16bf bf_ = load_b_bf16(vt, 256, dt * 16 + lm, nk, half); // B from LDS
      acc[dt] = wmma_bf16(af, bf_, acc[dt]);
    }
  }
#pragma unroll
  for (int dt = 0; dt < 8; ++dt) {
#pragma unroll
    for (int i = 0; i < 8; ++i) {
      atomicAdd(&kv[((size_t)b * C2 + c0 + 8 * half + i) * C2 + dt * 16 + lm], acc[dt][i]);
    }
  }
}

// ---- Stage 3a: Mtmp[b] = Wf * kv[b]^T  ([64x128]x[128x128], bf16 out) ----
__global__ __launch_bounds__(256) void mtmp_kernel(
    const float* __restrict__ Wf, const float* __restrict__ kv,
    bf16_t* __restrict__ mtmp) {
  const int b    = blockIdx.x;
  const int wave = threadIdx.x >> 5;
  const int lane = threadIdx.x & 31;
  const int half = lane >> 4, lm = lane & 15;
  const int mt   = wave & 3;
  const int cg   = (wave >> 2) * 4;
  const float* kvb = kv + (size_t)b * C2 * C2;

  v8f acc[4];
#pragma unroll
  for (int j = 0; j < 4; ++j) acc[j] = (v8f){};

#pragma unroll
  for (int d = 0; d < C2; d += 32) {
    v16bf af = load_a_f32(Wf, C2, mt * 16 + lm, d, C2, half);          // A = Wf (m=o, K=d)
#pragma unroll
    for (int j = 0; j < 4; ++j) {
      v16bf bf_ = load_b_f32_contig(kvb, C2, (cg + j) * 16 + lm, d, half); // B(k=d,n=c)=kv[c][d]
      acc[j] = wmma_bf16(af, bf_, acc[j]);
    }
  }
#pragma unroll
  for (int j = 0; j < 4; ++j) {
    int cidx = (cg + j) * 16 + lm;
#pragma unroll
    for (int i = 0; i < 8; ++i) {
      int o = mt * 16 + 8 * half + i;
      mtmp[((size_t)b * C_ + o) * C2 + cidx] = (bf16_t)acc[j][i];
    }
  }
}

// ---- Stage 3b: out = ReLU(BN(Mtmp[b] * q[b])); qT tile async-staged in LDS ----
// grid (N/16, B), 128 thr = 4 waves = 4 output-row tiles.
__global__ __launch_bounds__(128) void out_kernel(
    const bf16_t* __restrict__ mtmp, const bf16_t* __restrict__ qT,
    const float* __restrict__ sc, const float* __restrict__ sh,
    float* __restrict__ out) {
  __shared__ uint4 lds4[256];                  // qt: [n=16][c=128] bf16 = 4 KB
  bf16_t* qt = (bf16_t*)lds4;
  const int b  = blockIdx.y;
  const int n0 = blockIdx.x * 16;
  const int t  = threadIdx.x;
  const bf16_t* qb = qT + ((size_t)b * N_ + n0) * C2;

  const unsigned lbase = lds_offset(lds4);
#pragma unroll
  for (int i = 0; i < 2; ++i) {
    int id = t + i * 128;                      // id = row*16 + c8 (row stride = 16 chunks)
    async_copy_b128(lbase + (unsigned)id * 16,
                    qb + (size_t)(id >> 4) * C2 + (id & 15) * 8);
  }
  async_wait0();
  __syncthreads();

  const int wave = t >> 5, lane = t & 31;
  const int half = lane >> 4, lm = lane & 15;
  const int o0 = wave * 16;
  const bf16_t* mb = mtmp + (size_t)b * C_ * C2;

  v8f acc = {};
#pragma unroll
  for (int cB = 0; cB < C2; cB += 32) {
    v16bf af  = load_a_bf16(mb, C2, o0 + lm, cB, half);   // A = Mtmp (m=o, K=c)
    v16bf bf_ = load_b_bf16(qt, C2, lm, cB, half);        // B from LDS
    acc = wmma_bf16(af, bf_, acc);
  }
  const int n = n0 + lm;
#pragma unroll
  for (int i = 0; i < 8; ++i) {
    int o = o0 + 8 * half + i;
    out[((size_t)b * C_ + o) * N_ + n] =
        fmaxf(fmaf(acc[i], sc[3 * C2 + o], sh[3 * C2 + o]), 0.0f);
  }
}

extern "C" void kernel_launch(void* const* d_in, const int* in_sizes, int n_in,
                              void* d_out, int out_size, void* d_ws, size_t ws_size,
                              hipStream_t stream) {
  (void)in_sizes; (void)n_in; (void)out_size; (void)ws_size;
  const float* x   = (const float*)d_in[0];
  const float* Wq  = (const float*)d_in[1];
  const float* Wk  = (const float*)d_in[2];
  const float* Wv  = (const float*)d_in[3];
  const float* Wf  = (const float*)d_in[4];
  const float* bnq = (const float*)d_in[5];
  const float* bnk = (const float*)d_in[6];
  const float* bnv = (const float*)d_in[7];
  const float* bnf = (const float*)d_in[8];
  float* out = (float*)d_out;

  char* ws = (char*)d_ws;
  const size_t Sqkv = (size_t)B_ * C2 * N_ * sizeof(bf16_t);       // 32 MiB each
  const size_t SkvF = (size_t)B_ * C2 * C2 * sizeof(float);        // 512 KiB
  const size_t Smt  = (size_t)B_ * C_ * C2 * sizeof(bf16_t);       // 128 KiB
  const size_t Swp  = (size_t)C2 * KP * sizeof(bf16_t);            // 24 KiB each
  bf16_t* kbuf = (bf16_t*)(ws);
  bf16_t* vbuf = (bf16_t*)(ws + Sqkv);
  bf16_t* qT   = (bf16_t*)(ws + 2 * Sqkv);
  float*  kv   = (float*)(ws + 3 * Sqkv);
  bf16_t* mtmp = (bf16_t*)(ws + 3 * Sqkv + SkvF);
  bf16_t* Wqp  = (bf16_t*)(ws + 3 * Sqkv + SkvF + Smt);
  bf16_t* Wkp  = (bf16_t*)(ws + 3 * Sqkv + SkvF + Smt + Swp);
  bf16_t* Wvp  = (bf16_t*)(ws + 3 * Sqkv + SkvF + Smt + 2 * Swp);
  float*  sc   = (float*)(ws + 3 * Sqkv + SkvF + Smt + 3 * Swp);
  float*  sh   = sc + (3 * C2 + C_);

  const int prepN = 3 * C2 * KP + 3 * C2 + C_;
  prep_kernel<<<dim3((prepN + 255) / 256), 256, 0, stream>>>(
      Wq, Wk, Wv, bnq, bnk, bnv, bnf, Wqp, Wkp, Wvp, sc, sh);
  const int kvN = B_ * C2 * C2;
  zero_kernel<<<dim3((kvN + 255) / 256), 256, 0, stream>>>(kv, kvN);
  qkv_kernel<<<dim3(N_ / 16, B_), 256, 0, stream>>>(x, Wqp, Wkp, Wvp, sc, sh,
                                                    kbuf, vbuf, qT);
  kv_kernel<<<dim3(64, B_), 256, 0, stream>>>(kbuf, vbuf, kv);
  mtmp_kernel<<<dim3(B_), 256, 0, stream>>>(Wf, kv, mtmp);
  out_kernel<<<dim3(N_ / 16, B_), 128, 0, stream>>>(mtmp, qT, sc, sh, out);
}
